// Bayesian_GNN_node_Virtualnode_47081431499445
// MI455X (gfx1250) — compile-verified, matrix operand order checked by reading, not run on hardware
//
#include <hip/hip_runtime.h>
#include <stdint.h>

// Problem constants (fixed by the reference)
#define NNODES  100000
#define NEDGES  300000
#define DIM     256
#define TWO_D   512
#define NLAYERS 5
#define NGRAPH  4096

typedef __attribute__((ext_vector_type(16))) __bf16          v16bf;
typedef __attribute__((ext_vector_type(8)))  float           v8f;
typedef __attribute__((ext_vector_type(8)))  unsigned short  ushort8v;
typedef __attribute__((ext_vector_type(16))) unsigned short  ushort16v;
typedef __attribute__((ext_vector_type(4)))  unsigned short  ushort4v;

// ---------------- helpers ----------------

__device__ __forceinline__ unsigned short f2bf(float f) {
  unsigned int u = __builtin_bit_cast(unsigned int, f);
  u += 0x7FFFu + ((u >> 16) & 1u);            // round-to-nearest-even
  return (unsigned short)(u >> 16);
}

// Load one bf16 WMMA fragment half-row: 8 contiguous bf16 at p and 8 at p+16
// (matches 16-bit A/B layout: lane holds K = kSel..kSel+7 and kSel+16..kSel+23)
__device__ __forceinline__ v16bf load_frag(const unsigned short* p) {
  ushort8v lo = *(const ushort8v*)p;
  ushort8v hi = *(const ushort8v*)(p + 16);
  union { ushort16v v; ushort8v h[2]; } u;
  u.h[0] = lo; u.h[1] = hi;
  return __builtin_bit_cast(v16bf, u.v);
}

// ---------------- sparse / elementwise kernels ----------------

// h[n,:] = sum_f atom_emb[f, x[n,f], :]
__global__ void k_atom(const int* __restrict__ x, const float* __restrict__ emb,
                       float* __restrict__ h) {
  size_t tid = (size_t)blockIdx.x * blockDim.x + threadIdx.x;
  int n = (int)(tid >> 6);
  if (n >= NNODES) return;
  int d = ((int)tid & 63) * 4;
  float s0 = 0.f, s1 = 0.f, s2 = 0.f, s3 = 0.f;
#pragma unroll
  for (int f = 0; f < 9; ++f) {
    int idx = x[n * 9 + f];
    const float* p = emb + ((size_t)f * 120 + idx) * DIM + d;
    s0 += p[0]; s1 += p[1]; s2 += p[2]; s3 += p[3];
  }
  float* o = h + (size_t)n * DIM + d;
  o[0] = s0; o[1] = s1; o[2] = s2; o[3] = s3;
}

// broadcast vn0 -> vn[G, D]
__global__ void k_vn_init(const float* __restrict__ vn0, float* __restrict__ vn) {
  size_t tid = (size_t)blockIdx.x * blockDim.x + threadIdx.x;
  if (tid >= (size_t)NGRAPH * 64) return;
  int d = ((int)tid & 63) * 4;
  float* o = vn + tid * 4;
  o[0] = vn0[d]; o[1] = vn0[d + 1]; o[2] = vn0[d + 2]; o[3] = vn0[d + 3];
}

// transpose + convert weights: Wt[n*K + k] = bf16(W[k*Nout + n])
__global__ void k_wprep(const float* __restrict__ W, unsigned short* __restrict__ Wt,
                        int K, int Nout) {
  size_t tid = (size_t)blockIdx.x * blockDim.x + threadIdx.x;
  if (tid >= (size_t)K * Nout) return;
  int n = (int)(tid / K);
  int k = (int)(tid % K);
  Wt[tid] = f2bf(W[(size_t)k * Nout + n]);
}

// h += vn[batch]; aggr = 0; vtemp = 0
__global__ void k_add_vn(float* __restrict__ h, const float* __restrict__ vn,
                         const int* __restrict__ batch, float* __restrict__ aggr,
                         float* __restrict__ vtemp) {
  size_t tid = (size_t)blockIdx.x * blockDim.x + threadIdx.x;
  int n = (int)(tid >> 6);
  if (n >= NNODES) return;
  int d = ((int)tid & 63) * 4;
  int g = batch[n];
  float* hp = h + (size_t)n * DIM + d;
  const float* vp = vn + (size_t)g * DIM + d;
  hp[0] += vp[0]; hp[1] += vp[1]; hp[2] += vp[2]; hp[3] += vp[3];
  float* ap = aggr + (size_t)n * DIM + d;
  ap[0] = 0.f; ap[1] = 0.f; ap[2] = 0.f; ap[3] = 0.f;
  if (tid < (size_t)NGRAPH * 64) {
    float* tp = vtemp + tid * 4;
    tp[0] = 0.f; tp[1] = 0.f; tp[2] = 0.f; tp[3] = 0.f;
  }
}

// per edge: msg = relu(h[src] + bond_emb), scatter-add into aggr[dst]
__global__ void k_edge(const float* __restrict__ h, const int* __restrict__ ei,
                       const int* __restrict__ ea, const float* __restrict__ bond,
                       float* __restrict__ aggr, int layer) {
  size_t tid = (size_t)blockIdx.x * blockDim.x + threadIdx.x;
  int e = (int)(tid >> 6);
  if (e >= NEDGES) return;
  int d = ((int)tid & 63) * 4;
  int src = ei[e];
  int dst = ei[NEDGES + e];
  int a0 = ea[e * 3 + 0], a1 = ea[e * 3 + 1], a2 = ea[e * 3 + 2];
  const float* b0 = bond + (((size_t)layer * 3 + 0) * 6 + a0) * DIM + d;
  const float* b1 = bond + (((size_t)layer * 3 + 1) * 6 + a1) * DIM + d;
  const float* b2 = bond + (((size_t)layer * 3 + 2) * 6 + a2) * DIM + d;
  const float* hp = h + (size_t)src * DIM + d;
  float* ap = aggr + (size_t)dst * DIM + d;
#pragma unroll
  for (int i = 0; i < 4; ++i) {
    float m = fmaxf(hp[i] + b0[i] + b1[i] + b2[i], 0.0f);
    atomicAdd(ap + i, m);
  }
}

// t_bf16 = bf16((1+eps)*h + aggr); optional vtemp[batch[n]] += h[n]
__global__ void k_finalize(const float* __restrict__ h, const float* __restrict__ aggr,
                           const float* __restrict__ epsArr, int layer,
                           unsigned short* __restrict__ tbf, float* __restrict__ vtemp,
                           const int* __restrict__ batch, int doVn) {
  size_t tid = (size_t)blockIdx.x * blockDim.x + threadIdx.x;
  int n = (int)(tid >> 6);
  if (n >= NNODES) return;
  int d = ((int)tid & 63) * 4;
  float ep = 1.0f + epsArr[layer];
  const float* hp = h + (size_t)n * DIM + d;
  const float* ap = aggr + (size_t)n * DIM + d;
  ushort4v o;
#pragma unroll
  for (int i = 0; i < 4; ++i) o[i] = f2bf(ep * hp[i] + ap[i]);
  *(ushort4v*)(tbf + (size_t)n * DIM + d) = o;
  if (doVn) {
    int g = batch[n];
    float* vp = vtemp + (size_t)g * DIM + d;
#pragma unroll
    for (int i = 0; i < 4; ++i) atomicAdd(vp + i, hp[i]);
  }
}

// vt_bf16 = bf16(vtemp + vn)
__global__ void k_vn_pre(const float* __restrict__ vtemp, const float* __restrict__ vn,
                         unsigned short* __restrict__ vtbf) {
  size_t tid = (size_t)blockIdx.x * blockDim.x + threadIdx.x;
  if (tid >= (size_t)NGRAPH * 64) return;
  const float* a = vtemp + tid * 4;
  const float* b = vn + tid * 4;
  ushort4v o;
#pragma unroll
  for (int i = 0; i < 4; ++i) o[i] = f2bf(a[i] + b[i]);
  *(ushort4v*)(vtbf + tid * 4) = o;
}

// ---------------- WMMA GEMM + bias + eval-BN (+ReLU) epilogue ----------------
// C[M,Nout] = A[M,K](bf16) @ B[K,Nout]  with Bt = B^T stored [Nout,K] bf16.
// One wave -> 32(M) x 64(N) output tile (2 M-tiles x 4 N-tiles, 8 accumulators;
// each B fragment reused twice). 8 waves/block stacked along M (256 rows/block).
// M must be a multiple of 32 (100000 = 3125*32, 4096 = 128*32).
__global__ void k_gemm(const unsigned short* __restrict__ A,
                       const unsigned short* __restrict__ Bt,
                       const float* __restrict__ bias,
                       const float* __restrict__ bn,   // [4, Nout]: g,b,mean,var
                       float* __restrict__ outF,
                       unsigned short* __restrict__ outB,
                       int M, int K, int Nout, int doRelu) {
  int lane = threadIdx.x & 31;
  int wave = threadIdx.x >> 5;
  int pair = blockIdx.x * 8 + wave;            // 32 output rows per wave
  int mBase = pair * 32;
  if (mBase >= M) return;                      // uniform per wave: EXEC stays full
  int nBase = blockIdx.y * 64;

  int kSel = (lane & 16) ? 8 : 0;              // 16-bit A/B fragment K-half select
  const unsigned short* aRow0 = A + (size_t)(mBase + (lane & 15)) * K + kSel;
  const unsigned short* aRow1 = aRow0 + (size_t)16 * K;

  const unsigned short* bRow[4];
  int col[4];
#pragma unroll
  for (int j = 0; j < 4; ++j) {
    col[j] = nBase + j * 16 + (lane & 15);
    bRow[j] = Bt + (size_t)col[j] * K + kSel;
  }

  v8f zero = {0.f, 0.f, 0.f, 0.f, 0.f, 0.f, 0.f, 0.f};
  v8f acc0[4] = {zero, zero, zero, zero};
  v8f acc1[4] = {zero, zero, zero, zero};

  for (int kk = 0; kk < K; kk += 32) {
    __builtin_prefetch(aRow0 + kk + 64, 0, 3); // near-scope prefetch of next A chunk
    __builtin_prefetch(aRow1 + kk + 64, 0, 3);
    v16bf a0 = load_frag(aRow0 + kk);
    v16bf a1 = load_frag(aRow1 + kk);
#pragma unroll
    for (int j = 0; j < 4; ++j) {
      v16bf b = load_frag(bRow[j] + kk);
      acc0[j] = __builtin_amdgcn_wmma_f32_16x16x32_bf16(
          false, a0, false, b, (short)0, acc0[j], false, false);
      acc1[j] = __builtin_amdgcn_wmma_f32_16x16x32_bf16(
          false, a1, false, b, (short)0, acc1[j], false, false);
    }
  }

  // Epilogue. C layout: lane l, reg r -> M = r + ((l>=16)?8:0), N = l&15.
  int mOff = (lane & 16) ? 8 : 0;
#pragma unroll
  for (int j = 0; j < 4; ++j) {
    int c = col[j];
    float gamma = bn[c];
    float beta  = bn[Nout + c];
    float mean  = bn[2 * Nout + c];
    float var   = bn[3 * Nout + c];
    float scale = rsqrtf(var + 1e-5f) * gamma;
    float bi    = bias[c];
#pragma unroll
    for (int r = 0; r < 8; ++r) {
      float v0 = acc0[j][r] + bi;
      v0 = (v0 - mean) * scale + beta;
      float v1 = acc1[j][r] + bi;
      v1 = (v1 - mean) * scale + beta;
      if (doRelu) { v0 = fmaxf(v0, 0.0f); v1 = fmaxf(v1, 0.0f); }
      size_t i0 = (size_t)(mBase + mOff + r) * Nout + c;
      size_t i1 = (size_t)(mBase + 16 + mOff + r) * Nout + c;
      if (outB) { outB[i0] = f2bf(v0); outB[i1] = f2bf(v1); }
      else      { outF[i0] = v0;       outF[i1] = v1; }
    }
  }
}

// ---------------- host driver ----------------

extern "C" void kernel_launch(void* const* d_in, const int* in_sizes, int n_in,
                              void* d_out, int out_size, void* d_ws, size_t ws_size,
                              hipStream_t stream) {
  const int*   x        = (const int*)d_in[0];
  const int*   ei       = (const int*)d_in[1];
  const int*   ea       = (const int*)d_in[2];
  const int*   batch    = (const int*)d_in[3];
  const float* atom_emb = (const float*)d_in[4];
  const float* bond_emb = (const float*)d_in[5];
  const float* vn0      = (const float*)d_in[6];
  const float* eps      = (const float*)d_in[7];
  const float* cW1      = (const float*)d_in[8];
  const float* cb1      = (const float*)d_in[9];
  const float* cbn1     = (const float*)d_in[10];
  const float* cW2      = (const float*)d_in[11];
  const float* cb2      = (const float*)d_in[12];
  const float* nbn      = (const float*)d_in[13];
  const float* vW1      = (const float*)d_in[14];
  const float* vb1      = (const float*)d_in[15];
  const float* vbn1     = (const float*)d_in[16];
  const float* vW2      = (const float*)d_in[17];
  const float* vb2      = (const float*)d_in[18];
  const float* vbn2     = (const float*)d_in[19];

  // Workspace carve-out (256B aligned slices)
  char* wsp = (char*)d_ws;
  auto carve = [&](size_t bytes) -> void* {
    void* p = (void*)wsp;
    wsp += (bytes + 255) & ~(size_t)255;
    return p;
  };
  float*          h     = (float*)carve((size_t)NNODES * DIM * 4);
  float*          aggr  = (float*)carve((size_t)NNODES * DIM * 4);
  unsigned short* tbf   = (unsigned short*)carve((size_t)NNODES * DIM * 2);
  unsigned short* o1    = (unsigned short*)carve((size_t)NNODES * TWO_D * 2);
  float*          vtemp = (float*)carve((size_t)NGRAPH * DIM * 4);
  float*          vn    = (float*)carve((size_t)NGRAPH * DIM * 4);
  unsigned short* vtbf  = (unsigned short*)carve((size_t)NGRAPH * DIM * 2);
  unsigned short* ubf   = (unsigned short*)carve((size_t)NGRAPH * DIM * 2);
  unsigned short* W1t   = (unsigned short*)carve((size_t)NLAYERS * TWO_D * DIM * 2);
  unsigned short* W2t   = (unsigned short*)carve((size_t)NLAYERS * DIM * TWO_D * 2);
  unsigned short* vW1t  = (unsigned short*)carve((size_t)(NLAYERS - 1) * DIM * DIM * 2);
  unsigned short* vW2t  = (unsigned short*)carve((size_t)(NLAYERS - 1) * DIM * DIM * 2);

  dim3 blk(256);

  // One-time prep: weight transpose+bf16, atom encoder, vn broadcast
  for (int l = 0; l < NLAYERS; ++l) {
    k_wprep<<<(DIM * TWO_D + 255) / 256, blk, 0, stream>>>(
        cW1 + (size_t)l * DIM * TWO_D, W1t + (size_t)l * TWO_D * DIM, DIM, TWO_D);
    k_wprep<<<(TWO_D * DIM + 255) / 256, blk, 0, stream>>>(
        cW2 + (size_t)l * TWO_D * DIM, W2t + (size_t)l * DIM * TWO_D, TWO_D, DIM);
  }
  for (int l = 0; l < NLAYERS - 1; ++l) {
    k_wprep<<<(DIM * DIM + 255) / 256, blk, 0, stream>>>(
        vW1 + (size_t)l * DIM * DIM, vW1t + (size_t)l * DIM * DIM, DIM, DIM);
    k_wprep<<<(DIM * DIM + 255) / 256, blk, 0, stream>>>(
        vW2 + (size_t)l * DIM * DIM, vW2t + (size_t)l * DIM * DIM, DIM, DIM);
  }
  k_atom<<<(NNODES * 64) / 256, blk, 0, stream>>>(x, atom_emb, h);
  k_vn_init<<<(NGRAPH * 64) / 256, blk, 0, stream>>>(vn0, vn);

  const int mPairsN = NNODES / 32;                        // 3125
  const int mPairsG = NGRAPH / 32;                        // 128
  const dim3 gridN((mPairsN + 7) / 8, TWO_D / 64);        // 391 x 8
  const dim3 gridN2((mPairsN + 7) / 8, DIM / 64);         // 391 x 4
  const dim3 gridG(mPairsG / 8, DIM / 64);                // 16 x 4

  for (int l = 0; l < NLAYERS; ++l) {
    int doVn = (l < NLAYERS - 1) ? 1 : 0;

    k_add_vn<<<(NNODES * 64) / 256, blk, 0, stream>>>(h, vn, batch, aggr, vtemp);
    k_edge<<<(NEDGES * 64) / 256, blk, 0, stream>>>(h, ei, ea, bond_emb, aggr, l);
    k_finalize<<<(NNODES * 64) / 256, blk, 0, stream>>>(h, aggr, eps, l, tbf, vtemp,
                                                        batch, doVn);
    // GIN MLP: relu(BN1(t@W1+b1)) -> bf16
    k_gemm<<<gridN, blk, 0, stream>>>(tbf, W1t + (size_t)l * TWO_D * DIM,
                                      cb1 + (size_t)l * TWO_D,
                                      cbn1 + (size_t)l * 4 * TWO_D,
                                      nullptr, o1, NNODES, DIM, TWO_D, 1);
    // BN_node(o1@W2+b2) [+relu if not last] -> f32 into h (or d_out last layer)
    float* outPtr = (l == NLAYERS - 1) ? (float*)d_out : h;
    k_gemm<<<gridN2, blk, 0, stream>>>(o1, W2t + (size_t)l * DIM * TWO_D,
                                       cb2 + (size_t)l * DIM,
                                       nbn + (size_t)l * 4 * DIM,
                                       outPtr, nullptr, NNODES, TWO_D, DIM, doVn);
    if (doVn) {
      k_vn_pre<<<(NGRAPH * 64) / 256, blk, 0, stream>>>(vtemp, vn, vtbf);
      k_gemm<<<gridG, blk, 0, stream>>>(vtbf, vW1t + (size_t)l * DIM * DIM,
                                        vb1 + (size_t)l * DIM,
                                        vbn1 + (size_t)l * 4 * DIM,
                                        nullptr, ubf, NGRAPH, DIM, DIM, 1);
      k_gemm<<<gridG, blk, 0, stream>>>(ubf, vW2t + (size_t)l * DIM * DIM,
                                        vb2 + (size_t)l * DIM,
                                        vbn2 + (size_t)l * 4 * DIM,
                                        vn, nullptr, NGRAPH, DIM, DIM, 1);
    }
  }
}
